// Codebook_63969242907155
// MI455X (gfx1250) — matrix operand order repaired and checked by reading, not compiled, not run on hardware
//
#include <hip/hip_runtime.h>
#include <hip/hip_bf16.h>

typedef __attribute__((ext_vector_type(16))) _Float16 v16h;
typedef __attribute__((ext_vector_type(8)))  _Float16 v8h;
typedef __attribute__((ext_vector_type(8)))  float    v8f;

#define C_   256
#define HW_  4096
#define N_   131072
#define K_   1024
#define ZQ_ELEMS   8388608   // 32*256*64*64
#define GATHER_BLOCKS 32768  // ZQ_ELEMS / 256

// workspace byte offsets
#define WS_EBUF 0            // 1024*256 f16 = 524288 B
#define WS_NRM  524288       // 1024 f32    = 4096 B
#define WS_IDX  528384       // 131072 i32  = 524288 B
#define WS_PART 1052672      // 32768 f32   = 131072 B

// ---------------------------------------------------------------------------
// CDNA5 async global->LDS copy (ASYNCcnt-tracked) + wait, via inline asm
// (portable across ROCm7.2 / amdgpu-toolchain builtin-arity differences).
// LDS destination address = low 32 bits of the flat shared-pointer (ISA 10.2).
// ---------------------------------------------------------------------------
__device__ __forceinline__ void async_copy_b128(unsigned lds_off, const void* g) {
  asm volatile("global_load_async_to_lds_b128 %0, %1, off"
               :: "v"(lds_off), "v"((unsigned long long)(uintptr_t)g)
               : "memory");
}
__device__ __forceinline__ void wait_async0() {
  asm volatile("s_wait_asynccnt 0x0" ::: "memory");
}

// ---------------------------------------------------------------------------
// Prep: emb fp32 -> f16 swizzled into WMMA B-matrix lane layout, + ||e_k||^2.
// B operand as packed here (32x16 f16, contiguous k): lane L holds column n = L&15; lanes 0-15 hold
// K=0..15, lanes 16-31 hold K=16..31, as 16 consecutive halves (v16h).
// ebuf flat index: ((kt*8+cc)*32 + lane)*16 + j ,  c = cc*32 + 16*(L>>4) + j
// ---------------------------------------------------------------------------
__global__ __launch_bounds__(256)
void vq_prep(const float* __restrict__ emb, _Float16* __restrict__ ebuf,
             float* __restrict__ nrm) {
  const int k = blockIdx.x * 256 + threadIdx.x;      // 0..1023
  const float* er = emb + (size_t)k * C_;
  const int kt = k >> 4, kl = k & 15;
  float s = 0.f;
  for (int cc = 0; cc < 8; ++cc)
    for (int lh = 0; lh < 2; ++lh)
      for (int j = 0; j < 16; ++j) {
        const int c = cc * 32 + lh * 16 + j;
        ebuf[(((size_t)(kt * 8 + cc) * 32 + lh * 16 + kl) << 4) + j] = (_Float16)er[c];
      }
  for (int c = 0; c < 256; ++c) s += er[c] * er[c];
  nrm[k] = s;
}

// ---------------------------------------------------------------------------
// Argmin over d(n,k) = ||e_k||^2 - 2 z.e_k  (||z||^2 constant per row).
// Block: 256 threads = 8 waves, each wave owns 32 rows (two 16-row A tiles),
// block covers 256 rows. B panels (8KB = 16 columns x 256 c in f16) are
// double-buffered in LDS via async copies; each LDS B chunk feeds 2 WMMAs.
// A operand as loaded here (16x32 f16, wave32): lane L row M=L&15; lanes 0-15 hold K halves
// 0..15, lanes 16-31 hold 16..31 (contiguous, same order as B).
// D-matrix: VGPR v, lanes 0-15 -> (M=v, N=lane), lanes 16-31 -> (M=v+8).
// ---------------------------------------------------------------------------
__global__ __launch_bounds__(256)
void vq_argmin(const float* __restrict__ z, const _Float16* __restrict__ ebuf,
               const float* __restrict__ nrm, int* __restrict__ idx_i32,
               float* __restrict__ idx_f32) {
  __shared__ _Float16 As32[32][264];   // A staging, one 32-row round at a time
  __shared__ _Float16 Bs[2][4096];     // double-buffered B panel (8KB each)
  __shared__ float    nrmS[1024];

  const int tid  = threadIdx.x;
  const int wave = tid >> 5, lane = tid & 31;
  const int m    = lane & 15;
  const int hsel = (lane >> 4) << 4;                 // 0 or 16 (A half-lane K select)
  const long n0  = (long)blockIdx.x * 256;           // 4096 % 256 == 0: no batch cross
  const int b    = (int)(n0 >> 12);
  const int hw0  = (int)(n0 & 4095);
  const float* zb = z + (size_t)b * C_ * HW_ + hw0;  // zf[n0+r][c] = zb[c*4096 + r]

  // kick off async copy of B panel 0 (overlaps with A staging below)
  {
    const unsigned bs0 = (unsigned)(uintptr_t)&Bs[0][0];
    #pragma unroll
    for (int j = 0; j < 2; ++j) {
      const int e = tid + (j << 8);                  // 0..511, 16B each = 8KB
      async_copy_b128(bs0 + e * 16, ebuf + (size_t)e * 8);
    }
  }
  // stage codebook norms to LDS
  #pragma unroll
  for (int j = 0; j < 4; ++j) nrmS[tid + (j << 8)] = nrm[tid + (j << 8)];

  // stage A in 8 rounds of 32 rows; wave w builds its registers in round w
  v16h a0[8], a1[8];
  for (int w = 0; w < 8; ++w) {
    #pragma unroll 4
    for (int i = 0; i < 32; ++i) {
      const int e = tid + (i << 8);                  // 8192 elems: c = e/32, r = e%32
      const int c = e >> 5, r = e & 31;
      As32[r][c] = (_Float16)zb[(size_t)c * HW_ + (w << 5) + r];
    }
    __syncthreads();
    if (wave == w) {
      #pragma unroll
      for (int cc = 0; cc < 8; ++cc) {
        const _Float16* p0 = &As32[m][cc * 32];      // tile0: local rows 0..15
        const _Float16* p1 = &As32[16 + m][cc * 32]; // tile1: local rows 16..31
        const v8h lo0 = *(const v8h*)(p0 + hsel), hi0 = *(const v8h*)(p0 + 8 + hsel);
        const v8h lo1 = *(const v8h*)(p1 + hsel), hi1 = *(const v8h*)(p1 + 8 + hsel);
        #pragma unroll
        for (int j = 0; j < 8; ++j) {
          a0[cc][j] = lo0[j]; a0[cc][8 + j] = hi0[j];
          a1[cc][j] = lo1[j]; a1[cc][8 + j] = hi1[j];
        }
      }
    }
    __syncthreads();                                 // protect As32 reuse next round
  }

  float minv0[8], minv1[8]; int mini0[8], mini1[8];
  #pragma unroll
  for (int v = 0; v < 8; ++v) {
    minv0[v] = 3.4e38f; minv1[v] = 3.4e38f; mini0[v] = 0; mini1[v] = 0;
  }

  const unsigned bsb0 = (unsigned)(uintptr_t)&Bs[0][0];
  const unsigned bsb1 = (unsigned)(uintptr_t)&Bs[1][0];

  for (int kt = 0; kt < 64; ++kt) {
    wait_async0();                                   // my copy portion landed
    __syncthreads();                                 // everyone's portion + prev readers done
    if (kt < 63) {                                   // prefetch next panel into other buffer
      const int p = kt + 1;
      const _Float16* gp = ebuf + ((size_t)p << 12); // 4096 halves per panel
      const unsigned bl = (p & 1) ? bsb1 : bsb0;
      #pragma unroll
      for (int j = 0; j < 2; ++j) {
        const int e = tid + (j << 8);
        async_copy_b128(bl + e * 16, gp + (size_t)e * 8);
      }
    }
    const _Float16* bp = &Bs[kt & 1][0];
    v8f acc0 = {}, acc1 = {};
    #pragma unroll
    for (int cc = 0; cc < 8; ++cc) {
      const v16h bv = *(const v16h*)(bp + (cc << 9) + lane * 16);
      acc0 = __builtin_amdgcn_wmma_f32_16x16x32_f16(false, a0[cc], false, bv,
                                                    (short)0, acc0, false, false);
      acc1 = __builtin_amdgcn_wmma_f32_16x16x32_f16(false, a1[cc], false, bv,
                                                    (short)0, acc1, false, false);
    }
    const int   kk = (kt << 4) + m;                  // this lane's column k
    const float nl = nrmS[kk];
    #pragma unroll
    for (int v = 0; v < 8; ++v) {
      const float d0 = nl - 2.0f * acc0[v];
      if (d0 < minv0[v]) { minv0[v] = d0; mini0[v] = kk; }  // earlier kt wins ties
      const float d1 = nl - 2.0f * acc1[v];
      if (d1 < minv1[v]) { minv1[v] = d1; mini1[v] = kk; }
    }
  }

  // reduce across the 16 lanes of each half-wave (xor masks keep halves intact)
  #pragma unroll
  for (int off = 8; off >= 1; off >>= 1) {
    #pragma unroll
    for (int v = 0; v < 8; ++v) {
      float ov; int oi;
      ov = __shfl_xor(minv0[v], off, 32); oi = __shfl_xor(mini0[v], off, 32);
      if (ov < minv0[v] || (ov == minv0[v] && oi < mini0[v])) { minv0[v] = ov; mini0[v] = oi; }
      ov = __shfl_xor(minv1[v], off, 32); oi = __shfl_xor(mini1[v], off, 32);
      if (ov < minv1[v] || (ov == minv1[v] && oi < mini1[v])) { minv1[v] = ov; mini1[v] = oi; }
    }
  }

  if (m == 0) {                // lane 0 -> rows M=0..7, lane 16 -> M=8..15
    const long rb0 = n0 + (long)(wave << 5) + ((long)(lane >> 4) << 3);
    const long rb1 = rb0 + 16;
    #pragma unroll
    for (int v = 0; v < 8; ++v) {
      idx_i32[rb0 + v] = mini0[v]; idx_f32[rb0 + v] = (float)mini0[v];
      idx_i32[rb1 + v] = mini1[v]; idx_f32[rb1 + v] = (float)mini1[v];
    }
  }
}

// ---------------------------------------------------------------------------
// Gather zq = emb[idx] in [B,C,H,W] order + deterministic per-block loss partials
// ---------------------------------------------------------------------------
__global__ __launch_bounds__(256)
void vq_gather(const float* __restrict__ z, const float* __restrict__ emb,
               const int* __restrict__ idx_i32, float* __restrict__ outq,
               float* __restrict__ partials) {
  __shared__ float red[256];
  const int tid = threadIdx.x;
  const size_t o = (size_t)blockIdx.x * 256 + tid;   // flat [B,C,H,W] index
  const int hw = (int)(o & 4095);
  const int c  = (int)((o >> 12) & 255);
  const int b  = (int)(o >> 20);
  const int n  = (b << 12) + hw;
  const int id = idx_i32[n];
  const float q  = emb[(size_t)id * 256 + c];
  const float zv = z[o];
  outq[o] = q;                                       // straight-through fwd value == zq
  const float dd = q - zv;
  red[tid] = dd * dd;
  __syncthreads();
  for (int s = 128; s > 0; s >>= 1) {
    if (tid < s) red[tid] += red[tid + s];
    __syncthreads();
  }
  if (tid == 0) partials[blockIdx.x] = red[0];
}

__global__ __launch_bounds__(256)
void vq_loss(const float* __restrict__ partials, float* __restrict__ loss_out) {
  __shared__ float red[256];
  float s = 0.f;
  for (int i = threadIdx.x; i < GATHER_BLOCKS; i += 256) s += partials[i];
  red[threadIdx.x] = s;
  __syncthreads();
  for (int t = 128; t > 0; t >>= 1) {
    if (threadIdx.x < t) red[threadIdx.x] += red[threadIdx.x + t];
    __syncthreads();
  }
  // loss = mse(zq,z) - beta*mse(z,zq) = (1-0.25)*mean(diff^2)
  if (threadIdx.x == 0) *loss_out = 0.75f * red[0] / (float)ZQ_ELEMS;
}

// ---------------------------------------------------------------------------
extern "C" void kernel_launch(void* const* d_in, const int* in_sizes, int n_in,
                              void* d_out, int out_size, void* d_ws, size_t ws_size,
                              hipStream_t stream) {
  const float* z   = (const float*)d_in[0];   // [32,256,64,64]
  const float* emb = (const float*)d_in[1];   // [1024,256]
  float* out = (float*)d_out;
  char*  ws  = (char*)d_ws;

  _Float16* ebuf     = (_Float16*)(ws + WS_EBUF);
  float*    nrm      = (float*)(ws + WS_NRM);
  int*      idxi     = (int*)(ws + WS_IDX);
  float*    partials = (float*)(ws + WS_PART);

  float* zq    = out;                  // 8388608 floats, [B,C,H,W]
  float* idxf  = out + ZQ_ELEMS;       // 131072 indices (as float)
  float* lossp = out + ZQ_ELEMS + N_;  // scalar

  vq_prep  <<<K_ / 256,      256, 0, stream>>>(emb, ebuf, nrm);
  vq_argmin<<<N_ / 256,      256, 0, stream>>>(z, ebuf, nrm, idxi, idxf);
  vq_gather<<<GATHER_BLOCKS, 256, 0, stream>>>(z, emb, idxi, zq, partials);
  vq_loss  <<<1,             256, 0, stream>>>(partials, lossp);
}
